// AD_SUP2_MODEL7_22539988369914
// MI455X (gfx1250) — matrix operand order repaired and checked by reading, not compiled, not run on hardware
//
#include <hip/hip_runtime.h>
#include <hip/hip_bf16.h>
#include <math.h>

typedef __attribute__((ext_vector_type(16))) _Float16 v16h;
typedef __attribute__((ext_vector_type(8)))  _Float16 v8h;
typedef __attribute__((ext_vector_type(8)))  float    v8f;

#define TX    512
#define VSEQ  100
#define DIN   64
#define DMODEL 512
#define NHEAD 8
#define HDIM  64
#define DFF   2048
#define NLAYER 2
#define HLSTM 256
#define DATT  128
#define FCH   256
#define MROWS (VSEQ*TX)   /* 51200 */

// ---------------------------------------------------------------------------
// Generic mixed-precision WMMA GEMM:  out = act( A[M,K] * B[N,K]^T + bias + add )
// A,B are f16 row-major (B is the weight in torch [out,in] layout, i.e. already
// the transpose WMMA's B operand wants). One wave computes a 32x64 tile.
// Branch-free inner loop so the compiler keeps the 4x-unrolled, s_clause'd,
// partially-waited WMMA software pipeline.
// ---------------------------------------------------------------------------
__global__ __launch_bounds__(256) void gemm_wmma_kernel(
    const _Float16* __restrict__ A, int lda,
    const _Float16* __restrict__ B, int ldb,
    const float* __restrict__ bias,
    const float* __restrict__ addRow, int addRowDiv, int addLd,
    float* __restrict__ outF32, _Float16* __restrict__ outF16,
    int M, int N, int K, int act)
{
    const int lane = threadIdx.x & 31;
    const int wid  = threadIdx.x >> 5;
    const int gw   = blockIdx.x * 8 + wid;
    const int tilesN = N >> 6;
    const int wm = gw / tilesN;
    const int wn = gw - wm * tilesN;
    const int rowBase = wm << 5;
    const int colBase = wn << 6;
    if (rowBase >= M) return;          // uniform per wave: EXEC stays all-ones

    const int r = lane & 15;
    const int g = lane >> 4;

    v8f acc[2][4];
#pragma unroll
    for (int i = 0; i < 2; ++i)
#pragma unroll
        for (int j = 0; j < 4; ++j) acc[i][j] = (v8f){};

    // A fragment: lane holds row (rowBase + lane%16); K chunks at g*8 and 16+g*8.
    const _Float16* aBase0 = A + (size_t)(rowBase + r) * lda + g * 8;
    const _Float16* aBase1 = aBase0 + (size_t)16 * lda;
    // B fragment: lane holds W row (colBase + lane%16); K chunks at g*8 and 16+g*8.
    const _Float16* bBase  = B + (size_t)(colBase + r) * ldb + g * 8;

    for (int k = 0; k < K; k += 32) {
        v8h a0lo = *(const v8h*)(aBase0 + k);
        v8h a0hi = *(const v8h*)(aBase0 + k + 16);
        v8h a1lo = *(const v8h*)(aBase1 + k);
        v8h a1hi = *(const v8h*)(aBase1 + k + 16);
        v16h aF0 = __builtin_shufflevector(a0lo, a0hi, 0,1,2,3,4,5,6,7,8,9,10,11,12,13,14,15);
        v16h aF1 = __builtin_shufflevector(a1lo, a1hi, 0,1,2,3,4,5,6,7,8,9,10,11,12,13,14,15);
#pragma unroll
        for (int j = 0; j < 4; ++j) {
            const _Float16* bp = bBase + (size_t)(j * 16) * ldb + k;
            v8h blo = *(const v8h*)bp;
            v8h bhi = *(const v8h*)(bp + 16);
            v16h bF = __builtin_shufflevector(blo, bhi, 0,1,2,3,4,5,6,7,8,9,10,11,12,13,14,15);
            acc[0][j] = __builtin_amdgcn_wmma_f32_16x16x32_f16(
                false, aF0, false, bF, (short)0, acc[0][j], false, false);
            acc[1][j] = __builtin_amdgcn_wmma_f32_16x16x32_f16(
                false, aF1, false, bF, (short)0, acc[1][j], false, false);
        }
    }

    // Epilogue. C/D layout: lane L -> col L%16, VGPR r -> row (L/16)*8 + r.
    const int nl = lane & 15;
    const int mo = (lane >> 4) * 8;
#pragma unroll
    for (int i = 0; i < 2; ++i) {
#pragma unroll
        for (int j = 0; j < 4; ++j) {
            const int col = colBase + j * 16 + nl;
            const float bv = bias ? bias[col] : 0.0f;
#pragma unroll
            for (int rr = 0; rr < 8; ++rr) {
                const int row = rowBase + i * 16 + mo + rr;
                float v = acc[i][j][rr] + bv;
                if (addRow) v += addRow[(size_t)(row / addRowDiv) * addLd + col];
                if (act)    v = v > 0.0f ? v : 0.0f;
                const size_t oi = (size_t)row * N + col;
                if (outF32) outF32[oi] = v;
                if (outF16) outF16[oi] = (_Float16)v;
            }
        }
    }
}

// ---------------------------------------------------------------------------
// f32 -> f16 elementwise convert (grid-stride)
// ---------------------------------------------------------------------------
__global__ void cvt_kernel(const float* __restrict__ s, _Float16* __restrict__ d, size_t n)
{
    size_t i = (size_t)blockIdx.x * 256 + threadIdx.x;
    const size_t stride = (size_t)gridDim.x * 256;
    for (; i < n; i += stride) d[i] = (_Float16)s[i];
}

// x_data [Tx, V, D] f32  ->  XD [v*Tx+t, D] f16  (the swapaxes + cvt)
__global__ void xd_kernel(const float* __restrict__ x, _Float16* __restrict__ dst)
{
    const size_t idx = (size_t)blockIdx.x * 256 + threadIdx.x;
    const size_t total = (size_t)MROWS * DIN;
    if (idx >= total) return;
    const int m = (int)(idx / DIN);
    const int k = (int)(idx - (size_t)m * DIN);
    const int v = m / TX;
    const int t = m - v * TX;
    dst[idx] = (_Float16)x[((size_t)(t * VSEQ + v)) * DIN + k];
}

// sinusoidal positional encoding, pe[v, d]
__global__ void pe_kernel(float* __restrict__ pe)
{
    const int idx = blockIdx.x * 256 + threadIdx.x;
    if (idx >= VSEQ * DMODEL) return;
    const int v = idx / DMODEL;
    const int d = idx - v * DMODEL;
    const int j = d >> 1;
    const float freq = expf((float)(2 * j) * (-9.210340371976184f / (float)DMODEL));
    const float ang = (float)v * freq;
    pe[idx] = (d & 1) ? cosf(ang) : sinf(ang);
}

// ---------------------------------------------------------------------------
// LayerNorm(x + res) * g + b  ->  out f32 + f16 mirror. One block per row.
// ---------------------------------------------------------------------------
__global__ __launch_bounds__(256) void ln_kernel(
    const float* __restrict__ x, const float* __restrict__ res,
    const float* __restrict__ gam, const float* __restrict__ bet,
    float* __restrict__ o32, _Float16* __restrict__ o16)
{
    __shared__ float s1[256], s2[256];
    const int row = blockIdx.x;
    const int tid = threadIdx.x;
    const size_t base = (size_t)row * DMODEL;
    const float v0 = x[base + tid]       + res[base + tid];
    const float v1 = x[base + tid + 256] + res[base + tid + 256];
    s1[tid] = v0 + v1;
    s2[tid] = v0 * v0 + v1 * v1;
    __syncthreads();
    for (int off = 128; off > 0; off >>= 1) {
        if (tid < off) { s1[tid] += s1[tid + off]; s2[tid] += s2[tid + off]; }
        __syncthreads();
    }
    const float mean = s1[0] * (1.0f / DMODEL);
    const float var  = s2[0] * (1.0f / DMODEL) - mean * mean;
    const float rstd = rsqrtf(var + 1e-5f);
    const float y0 = (v0 - mean) * rstd * gam[tid]       + bet[tid];
    const float y1 = (v1 - mean) * rstd * gam[tid + 256] + bet[tid + 256];
    o32[base + tid] = y0;           o32[base + tid + 256] = y1;
    o16[base + tid] = (_Float16)y0; o16[base + tid + 256] = (_Float16)y1;
}

// ---------------------------------------------------------------------------
// Multi-head self attention over the V=100 axis. One block per (t, head).
// qkv f16 [v*Tx+t, 3*DM]; ctx f16 [v*Tx+t, DM].
// ---------------------------------------------------------------------------
__global__ __launch_bounds__(256) void attn_kernel(
    const _Float16* __restrict__ qkv, _Float16* __restrict__ ctx)
{
    __shared__ _Float16 qs[VSEQ * HDIM], ks[VSEQ * HDIM], vs[VSEQ * HDIM];
    __shared__ float sc[32 * VSEQ];
    const int t   = blockIdx.x >> 3;
    const int h   = blockIdx.x & 7;
    const int tid = threadIdx.x;

    for (int idx = tid; idx < VSEQ * HDIM; idx += 256) {
        const int vv = idx / HDIM;
        const int d  = idx - vv * HDIM;
        const size_t base = ((size_t)(vv * TX + t)) * (3 * DMODEL) + h * HDIM + d;
        qs[idx] = qkv[base];
        ks[idx] = qkv[base + DMODEL];
        vs[idx] = qkv[base + 2 * DMODEL];
    }
    __syncthreads();

    for (int i0 = 0; i0 < VSEQ; i0 += 32) {
        const int rows = (VSEQ - i0) < 32 ? (VSEQ - i0) : 32;
        for (int e = tid; e < rows * VSEQ; e += 256) {
            const int i = e / VSEQ;
            const int j = e - i * VSEQ;
            float s = 0.0f;
            const _Float16* qp = &qs[(i0 + i) * HDIM];
            const _Float16* kp = &ks[j * HDIM];
            for (int d = 0; d < HDIM; ++d) s += (float)qp[d] * (float)kp[d];
            sc[e] = s * 0.125f;   // 1/sqrt(64)
        }
        __syncthreads();
        if (tid < rows) {
            float* rowp = &sc[tid * VSEQ];
            float mx = rowp[0];
            for (int j = 1; j < VSEQ; ++j) mx = fmaxf(mx, rowp[j]);
            float sum = 0.0f;
            for (int j = 0; j < VSEQ; ++j) { const float e = expf(rowp[j] - mx); rowp[j] = e; sum += e; }
            const float inv = 1.0f / sum;
            for (int j = 0; j < VSEQ; ++j) rowp[j] *= inv;
        }
        __syncthreads();
        for (int e = tid; e < rows * HDIM; e += 256) {
            const int i = e / HDIM;
            const int d = e - i * HDIM;
            float s = 0.0f;
            const float* ap = &sc[i * VSEQ];
            for (int j = 0; j < VSEQ; ++j) s += ap[j] * (float)vs[j * HDIM + d];
            ctx[((size_t)((i0 + i) * TX + t)) * DMODEL + h * HDIM + d] = (_Float16)s;
        }
        __syncthreads();
    }
}

// ---------------------------------------------------------------------------
// Sequential attention-LSTM decoder: one 1024-thread workgroup (32 waves on a
// WGP), 512 steps. pre[v*Tx+t, DATT] = att1_b + h @ att1_w[:, :DM]^T was
// hoisted into a WMMA GEMM; only the state-dependent parts remain here.
// Only the final step's FC output is needed (log-probs don't feed the carry).
// ---------------------------------------------------------------------------
__global__ __launch_bounds__(1024) void decoder_kernel(
    const float* __restrict__ hs,
    const float* __restrict__ pre,
    const float* __restrict__ att1_w,
    const float* __restrict__ att2_w, const float* __restrict__ att2_b,
    const float* __restrict__ wih, const float* __restrict__ whh,
    const float* __restrict__ bih, const float* __restrict__ bhh,
    const float* __restrict__ fc1_w, const float* __restrict__ fc1_b,
    const float* __restrict__ fc2_w, const float* __restrict__ fc2_b,
    float* __restrict__ out)
{
    __shared__ float sh[HLSTM], scs[HLSTM], w2[DATT], a2[128], alf[128];
    __shared__ float cvec[DMODEL], cpart[1024], gates[4 * HLSTM], f1[FCH];
    __shared__ float red1[128];
    const int tid = threadIdx.x;
    if (tid < HLSTM) { sh[tid] = 0.0f; scs[tid] = 0.0f; }
    __syncthreads();

    for (int t = 0; t < TX; ++t) {
        // state-dependent half of att1: w2[j] = att1_w[j, DM:] . sh  (2 thr/out)
        if (tid < 2 * DATT) {
            const int j = tid >> 1, half = tid & 1;
            const float* wp = att1_w + (size_t)j * (DMODEL + HLSTM) + DMODEL + half * 128;
            float s = 0.0f;
            for (int d = 0; d < 128; ++d) s += wp[d] * sh[half * 128 + d];
            s += __shfl_down(s, 1, 2);
            if (half == 0) w2[j] = s;
        }
        __syncthreads();
        // a2[row] = att2_w . tanh(pre[row] + w2) + att2_b  (8 threads per row)
        {
            const int row = tid >> 3, sub = tid & 7;
            float s = 0.0f;
            if (row < VSEQ) {
                const float* pp = pre + ((size_t)(row * TX + t)) * DATT + sub * 16;
                const float* aw = att2_w + sub * 16;
                const float* wv = w2 + sub * 16;
                for (int j = 0; j < 16; ++j) s += aw[j] * tanhf(pp[j] + wv[j]);
            }
            s += __shfl_down(s, 4, 8);
            s += __shfl_down(s, 2, 8);
            s += __shfl_down(s, 1, 8);
            if (sub == 0 && row < VSEQ) a2[row] = s + att2_b[0];
        }
        __syncthreads();
        // parallel softmax over V
        if (tid < 128) red1[tid] = (tid < VSEQ) ? a2[tid] : -3.0e38f;
        __syncthreads();
        for (int off2 = 64; off2 > 0; off2 >>= 1) {
            if (tid < off2) red1[tid] = fmaxf(red1[tid], red1[tid + off2]);
            __syncthreads();
        }
        const float mx = red1[0];
        __syncthreads();
        if (tid < 128) { const float e = (tid < VSEQ) ? expf(a2[tid] - mx) : 0.0f;
                         alf[tid] = e; red1[tid] = e; }
        __syncthreads();
        for (int off2 = 64; off2 > 0; off2 >>= 1) {
            if (tid < off2) red1[tid] += red1[tid + off2];
            __syncthreads();
        }
        const float inv = 1.0f / (red1[0] + 1e-15f);
        // context c[d] = inv * sum_v alf[v] * hs[v,t,d]   (2 threads per d)
        {
            const int d = tid & 511, half = tid >> 9;
            float s = 0.0f;
            for (int v = half * 50; v < half * 50 + 50; ++v)
                s += alf[v] * hs[((size_t)(v * TX + t)) * DMODEL + d];
            cpart[tid] = s;
        }
        __syncthreads();
        if (tid < DMODEL) cvec[tid] = (cpart[tid] + cpart[tid + 512]) * inv;
        __syncthreads();
        // LSTM gates: exactly one gate per thread (uses previous hidden state)
        {
            float s = bih[tid] + bhh[tid];
            const float* wi = wih + (size_t)tid * DMODEL;
            for (int d = 0; d < DMODEL; ++d) s += wi[d] * cvec[d];
            const float* wh = whh + (size_t)tid * HLSTM;
            for (int d = 0; d < HLSTM; ++d) s += wh[d] * sh[d];
            gates[tid] = s;
        }
        __syncthreads();
        if (t == TX - 1) {   // only last step's log-probs are returned
            if (tid < FCH) {
                float s = fc1_b[tid];
                const float* wp = fc1_w + (size_t)tid * (DMODEL + HLSTM);
                for (int d = 0; d < DMODEL; ++d) s += wp[d] * cvec[d];
                for (int d = 0; d < HLSTM; ++d)  s += wp[DMODEL + d] * sh[d];
                f1[tid] = s > 0.0f ? s : 0.0f;
            }
            __syncthreads();
            if (tid == 0) {
                float z0 = fc2_b[0], z1 = fc2_b[1];
                for (int d = 0; d < FCH; ++d) { z0 += fc2_w[d] * f1[d]; z1 += fc2_w[FCH + d] * f1[d]; }
                const float m = fmaxf(z0, z1);
                const float lse = m + logf(expf(z0 - m) + expf(z1 - m));
                out[0] = z0 - lse;
                out[1] = z1 - lse;
            }
            __syncthreads();
        }
        // state update
        if (tid < HLSTM) {
            const float ig = gates[tid], fg = gates[HLSTM + tid];
            const float gg = gates[2 * HLSTM + tid], og = gates[3 * HLSTM + tid];
            const float si = 1.0f / (1.0f + expf(-ig));
            const float sf = 1.0f / (1.0f + expf(-fg));
            const float so = 1.0f / (1.0f + expf(-og));
            const float nc = sf * scs[tid] + si * tanhf(gg);
            scs[tid] = nc;
            sh[tid]  = so * tanhf(nc);
        }
        __syncthreads();
    }
}

// ---------------------------------------------------------------------------
extern "C" void kernel_launch(void* const* d_in, const int* in_sizes, int n_in,
                              void* d_out, int out_size, void* d_ws, size_t ws_size,
                              hipStream_t stream)
{
    (void)in_sizes; (void)n_in; (void)out_size; (void)ws_size;

    const float* x_data  = (const float*)d_in[0];
    const float* feat_w  = (const float*)d_in[1];
    const float* feat_b  = (const float*)d_in[2];
    const float* in_w    = (const float*)d_in[3];
    const float* in_b    = (const float*)d_in[4];
    const float* out_w   = (const float*)d_in[5];
    const float* out_b   = (const float*)d_in[6];
    const float* ff1_w   = (const float*)d_in[7];
    const float* ff1_b   = (const float*)d_in[8];
    const float* ff2_w   = (const float*)d_in[9];
    const float* ff2_b   = (const float*)d_in[10];
    const float* ln1_g   = (const float*)d_in[11];
    const float* ln1_b   = (const float*)d_in[12];
    const float* ln2_g   = (const float*)d_in[13];
    const float* ln2_b   = (const float*)d_in[14];
    const float* att1_w  = (const float*)d_in[15];
    const float* att1_b  = (const float*)d_in[16];
    const float* att2_w  = (const float*)d_in[17];
    const float* att2_b  = (const float*)d_in[18];
    const float* lstm_wih = (const float*)d_in[19];
    const float* lstm_whh = (const float*)d_in[20];
    const float* lstm_bih = (const float*)d_in[21];
    const float* lstm_bhh = (const float*)d_in[22];
    const float* fc1_w   = (const float*)d_in[23];
    const float* fc1_b   = (const float*)d_in[24];
    const float* fc2_w   = (const float*)d_in[25];
    const float* fc2_b   = (const float*)d_in[26];

    char* ws = (char*)d_ws;
    size_t off = 0;
    auto alloc = [&](size_t bytes) -> char* {
        char* p = ws + off;
        off += (bytes + 255) & ~(size_t)255;
        return p;
    };

    float*    A32   = (float*)   alloc((size_t)MROWS * DMODEL * 4);  // activations f32
    float*    T32   = (float*)   alloc((size_t)MROWS * DMODEL * 4);  // proj/ff2 tmp, later att1pre
    _Float16* A16   = (_Float16*)alloc((size_t)MROWS * DMODEL * 2);  // f16 mirror of activations
    _Float16* U16   = (_Float16*)alloc((size_t)MROWS * DFF    * 2);  // QKV / FF1 union (time-disjoint)
    _Float16* C16   = (_Float16*)alloc((size_t)MROWS * DMODEL * 2);  // attention context f16
    _Float16* XD16  = (_Float16*)alloc((size_t)MROWS * DIN    * 2);  // transposed input f16
    float*    PE32  = (float*)   alloc((size_t)VSEQ  * DMODEL * 4);  // positional encoding
    _Float16* fw16  = (_Float16*)alloc((size_t)DMODEL * DIN * 2);
    _Float16* iw16  = (_Float16*)alloc((size_t)NLAYER * 3 * DMODEL * DMODEL * 2);
    _Float16* ow16  = (_Float16*)alloc((size_t)NLAYER * DMODEL * DMODEL * 2);
    _Float16* f1w16 = (_Float16*)alloc((size_t)NLAYER * DFF * DMODEL * 2);
    _Float16* f2w16 = (_Float16*)alloc((size_t)NLAYER * DMODEL * DFF * 2);
    _Float16* aw16  = (_Float16*)alloc((size_t)DATT * (DMODEL + HLSTM) * 2);

    auto cvt = [&](const float* s, _Float16* d, size_t n) {
        const int blocks = (int)((n + 255) / 256);
        cvt_kernel<<<blocks, 256, 0, stream>>>(s, d, n);
    };
    auto gemm = [&](const _Float16* A, int lda, const _Float16* B, int ldb,
                    const float* bias, const float* addRow, int addDiv, int addLd,
                    float* o32, _Float16* o16, int M, int N, int K, int act) {
        const int blocks = (M / 32) * (N / 64) / 8;
        gemm_wmma_kernel<<<blocks, 256, 0, stream>>>(A, lda, B, ldb, bias,
                                                     addRow, addDiv, addLd,
                                                     o32, o16, M, N, K, act);
    };

    // weight conversions + input transpose + positional encoding
    cvt(feat_w, fw16,  (size_t)DMODEL * DIN);
    cvt(in_w,   iw16,  (size_t)NLAYER * 3 * DMODEL * DMODEL);
    cvt(out_w,  ow16,  (size_t)NLAYER * DMODEL * DMODEL);
    cvt(ff1_w,  f1w16, (size_t)NLAYER * DFF * DMODEL);
    cvt(ff2_w,  f2w16, (size_t)NLAYER * DMODEL * DFF);
    cvt(att1_w, aw16,  (size_t)DATT * (DMODEL + HLSTM));
    xd_kernel<<<(int)(((size_t)MROWS * DIN + 255) / 256), 256, 0, stream>>>(x_data, XD16);
    pe_kernel<<<(VSEQ * DMODEL + 255) / 256, 256, 0, stream>>>(PE32);

    // feature mapping + PE:  x = xd @ feat_w^T + feat_b + pe[v]
    gemm(XD16, DIN, fw16, DIN, feat_b, PE32, TX, DMODEL,
         A32, A16, MROWS, DMODEL, DIN, 0);

    for (int l = 0; l < NLAYER; ++l) {
        // QKV
        gemm(A16, DMODEL, iw16 + (size_t)l * 3 * DMODEL * DMODEL, DMODEL,
             in_b + (size_t)l * 3 * DMODEL, nullptr, 1, 0,
             nullptr, U16, MROWS, 3 * DMODEL, DMODEL, 0);
        // attention
        attn_kernel<<<TX * NHEAD, 256, 0, stream>>>(U16, C16);
        // output projection -> T32
        gemm(C16, DMODEL, ow16 + (size_t)l * DMODEL * DMODEL, DMODEL,
             out_b + (size_t)l * DMODEL, nullptr, 1, 0,
             T32, nullptr, MROWS, DMODEL, DMODEL, 0);
        // x = LN(x + proj)
        ln_kernel<<<MROWS, 256, 0, stream>>>(A32, T32,
            ln1_g + (size_t)l * DMODEL, ln1_b + (size_t)l * DMODEL, A32, A16);
        // FF1 (relu) -> U16
        gemm(A16, DMODEL, f1w16 + (size_t)l * DFF * DMODEL, DMODEL,
             ff1_b + (size_t)l * DFF, nullptr, 1, 0,
             nullptr, U16, MROWS, DFF, DMODEL, 1);
        // FF2 -> T32
        gemm(U16, DFF, f2w16 + (size_t)l * DMODEL * DFF, DFF,
             ff2_b + (size_t)l * DMODEL, nullptr, 1, 0,
             T32, nullptr, MROWS, DMODEL, DFF, 0);
        // x = LN(x + ff)
        ln_kernel<<<MROWS, 256, 0, stream>>>(A32, T32,
            ln2_g + (size_t)l * DMODEL, ln2_b + (size_t)l * DMODEL, A32, A16);
    }

    // hoisted state-independent half of decoder attention:
    // pre = hs @ att1_w[:, :DM]^T + att1_b   (stored [MROWS, DATT] in T32)
    gemm(A16, DMODEL, aw16, DMODEL + HLSTM, att1_b, nullptr, 1, 0,
         T32, nullptr, MROWS, DATT, DMODEL, 0);

    // sequential decoder (single 1024-thread workgroup)
    decoder_kernel<<<1, 1024, 0, stream>>>(A32, T32, att1_w, att2_w, att2_b,
                                           lstm_wih, lstm_whh, lstm_bih, lstm_bhh,
                                           fc1_w, fc1_b, fc2_w, fc2_b,
                                           (float*)d_out);
}
